// Policy_53626961657857
// MI455X (gfx1250) — matrix operand order repaired and checked by reading, not compiled, not run on hardware
//
#include <hip/hip_runtime.h>
#include <math.h>

typedef __attribute__((ext_vector_type(16))) __bf16 v16bf;
typedef __attribute__((ext_vector_type(8)))  float  v8f;

#define NF 256   // input feature dim (per half of concat)
#define L0 256
#define L1 128

__device__ __forceinline__ unsigned short f2bf(float f) {
    // round-to-nearest-even f32 -> bf16
    unsigned u = __float_as_uint(f);
    unsigned r = 0x7fffu + ((u >> 16) & 1u);
    return (unsigned short)((u + r) >> 16);
}

// ---------------- init: scores = -1e30 (softmax sentinel), counter = 0 ----------------
__global__ void k_init(float* __restrict__ scores, int* __restrict__ counter, int M) {
    int i = blockIdx.x * blockDim.x + threadIdx.x;
    if (i < M) scores[i] = -1e30f;
    if (i == 0) *counter = 0;
}

// ---------------- compact surviving node indices (mask = adj!=0, plus sentinel v==N) ----------------
__global__ void k_compact(const int* __restrict__ adj, int* __restrict__ list,
                          int* __restrict__ counter, int N) {
    int v = blockIdx.x * blockDim.x + threadIdx.x;
    if (v > N) return;
    bool m = (v == N) || (adj[v] != 0);
    if (m) {
        int p = atomicAdd(counter, 1);
        list[p] = v;
    }
}

// ---------------- c0[j] = b0[j] + sum_k feature[0][k] * W0[k][j]  (self-feature term) ----------------
__global__ void k_c0(const float* __restrict__ feat, const float* __restrict__ W0,
                     const float* __restrict__ b0, float* __restrict__ c0) {
    int j = threadIdx.x;            // 256 threads
    float s = b0[j];
    for (int k = 0; k < NF; ++k) s = fmaf(feat[k], W0[k * L0 + j], s);
    c0[j] = s;
}

// ---------------- pack W0b (rows 256..511 of W0) into WMMA B-operand layout, bf16 ----------------
// Bpack index = (((kt*16 + nt)*32 + lane)*16 + e)
// lanes 0-15: K = kt*32 + e      (e<8), kt*32 + 8+e (e>=8), column n = nt*16 + lane
// lanes 16-31: K = kt*32 + 8+e (e<8), kt*32 + 16+e (e>=8),            column n = nt*16 + (lane-16)
__global__ void k_pack0(const float* __restrict__ W0, unsigned short* __restrict__ P) {
    int idx = blockIdx.x * blockDim.x + threadIdx.x;    // 65536 total
    int e    = idx & 15;
    int lane = (idx >> 4) & 31;
    int nt   = (idx >> 9) & 15;
    int kt   = idx >> 13;
    int kl   = (e < 8) ? ((lane >> 4) * 8 + e) : (16 + (lane >> 4) * 8 + (e - 8));
    int k    = kt * 32 + kl;
    int n    = nt * 16 + (lane & 15);
    P[idx] = f2bf(W0[(NF + k) * L0 + n]);
}

__global__ void k_pack1(const float* __restrict__ W1, unsigned short* __restrict__ P) {
    int idx = blockIdx.x * blockDim.x + threadIdx.x;    // 32768 total
    int e    = idx & 15;
    int lane = (idx >> 4) & 31;
    int nt   = (idx >> 9) & 7;
    int kt   = idx >> 12;
    int kl   = (e < 8) ? ((lane >> 4) * 8 + e) : (16 + (lane >> 4) * 8 + (e - 8));
    int k    = kt * 32 + kl;
    int n    = nt * 16 + (lane & 15);
    P[idx] = f2bf(W1[k * L1 + n]);
}

// ---------------- main MLP: one wave per 16-row tile of compacted node list ----------------
__global__ __launch_bounds__(256) void k_mlp(
    const float* __restrict__ feat,
    const float* __restrict__ c0,
    const float* __restrict__ b1,
    const float* __restrict__ wsc,
    const unsigned short* __restrict__ P0,
    const unsigned short* __restrict__ P1,
    const int* __restrict__ list,
    const int* __restrict__ counter,
    float* __restrict__ scores,
    int N)
{
    __shared__ __align__(16) unsigned short lds_h1[8 * 16 * L0];   // 64 KB: 8 waves x (16 x 256) bf16

    const int lane = threadIdx.x & 31;
    const int wave = threadIdx.x >> 5;
    const int cnt  = *counter;
    const int tile = blockIdx.x * 8 + wave;
    const int base = tile * 16;
    if (base >= cnt) return;     // wave-local exit; no block barriers used below

    const int hh = lane >> 4;    // half-wave: 0 or 1
    const int lm = lane & 15;

    // A-operand row for this lane (16-bit A layout: lanes 0-15 and 16-31 both cover M=0..15)
    const int slotA   = base + lm;
    const int vA      = (slotA < cnt) ? list[slotA] : -1;
    const bool validA = (vA >= 1) && (vA < N);    // v==0 and sentinel get zero neighbor feature
    const float* rowp = feat + (long long)(validA ? vA : 0) * NF;

    // ---------------- layer 1: h1 = relu(x @ W0b + c0), M=16 K=256 N=256 ----------------
    v8f acc[16];
    #pragma unroll
    for (int nt = 0; nt < 16; ++nt)
        #pragma unroll
        for (int i = 0; i < 8; ++i) acc[nt][i] = 0.0f;

    #pragma unroll
    for (int kt = 0; kt < 8; ++kt) {
        // A layout: lanes 0-15 elems 0..7 -> K=kt*32+0..7, elems 8..15 -> K=kt*32+16..23
        //           lanes 16-31 shifted by +8
        const int k0 = kt * 32 + hh * 8;
        union { unsigned short h[16]; v16bf v; } A;
        float4 f0 = {0,0,0,0}, f1 = {0,0,0,0}, f2 = {0,0,0,0}, f3 = {0,0,0,0};
        if (validA) {
            f0 = *(const float4*)(rowp + k0);
            f1 = *(const float4*)(rowp + k0 + 4);
            f2 = *(const float4*)(rowp + k0 + 16);
            f3 = *(const float4*)(rowp + k0 + 20);
        }
        A.h[0]=f2bf(f0.x); A.h[1]=f2bf(f0.y); A.h[2]=f2bf(f0.z); A.h[3]=f2bf(f0.w);
        A.h[4]=f2bf(f1.x); A.h[5]=f2bf(f1.y); A.h[6]=f2bf(f1.z); A.h[7]=f2bf(f1.w);
        A.h[8]=f2bf(f2.x); A.h[9]=f2bf(f2.y); A.h[10]=f2bf(f2.z); A.h[11]=f2bf(f2.w);
        A.h[12]=f2bf(f3.x); A.h[13]=f2bf(f3.y); A.h[14]=f2bf(f3.z); A.h[15]=f2bf(f3.w);

        if (kt < 7) __builtin_prefetch(P0 + ((((kt + 1) * 16) * 32 + lane) << 4), 0, 0);

        #pragma unroll
        for (int nt = 0; nt < 16; ++nt) {
            const v16bf B = *(const v16bf*)(P0 + (((kt * 16 + nt) * 32 + lane) << 4));
            acc[nt] = __builtin_amdgcn_wmma_f32_16x16x32_bf16(
                false, A.v, false, B, (short)0, acc[nt], false, false);
        }
    }

    // bias + relu, write bf16 to LDS in row-major [m][n] (C layout -> staging for layer-2 A)
    unsigned short* myl = lds_h1 + wave * (16 * L0);
    #pragma unroll
    for (int nt = 0; nt < 16; ++nt) {
        const int n = nt * 16 + lm;
        const float cc = c0[n];
        #pragma unroll
        for (int r = 0; r < 8; ++r) {
            const int m = r + hh * 8;             // C layout: VGPR r, halves hold M=r / M=8+r
            float h = fmaxf(acc[nt][r] + cc, 0.0f);
            myl[m * L0 + n] = f2bf(h);
        }
    }
    asm volatile("s_wait_dscnt 0" ::: "memory");   // wave-local LDS RAW fence

    // ---------------- layer 2: h2 = relu(h1 @ W1 + b1), M=16 K=256 N=128 ----------------
    v8f acc2[8];
    #pragma unroll
    for (int nt = 0; nt < 8; ++nt)
        #pragma unroll
        for (int i = 0; i < 8; ++i) acc2[nt][i] = 0.0f;

    const unsigned short* arow = myl + lm * L0;    // this lane's A row (M=lm)
    #pragma unroll
    for (int kt = 0; kt < 8; ++kt) {
        const int k0 = kt * 32 + hh * 8;
        union { uint4 q[2]; v16bf v; } A;
        A.q[0] = *(const uint4*)(arow + k0);        // ds_load_b128: K = k0..k0+7
        A.q[1] = *(const uint4*)(arow + k0 + 16);   // ds_load_b128: K = k0+16..k0+23
        #pragma unroll
        for (int nt = 0; nt < 8; ++nt) {
            const v16bf B = *(const v16bf*)(P1 + (((kt * 8 + nt) * 32 + lane) << 4));
            acc2[nt] = __builtin_amdgcn_wmma_f32_16x16x32_bf16(
                false, A.v, false, B, (short)0, acc2[nt], false, false);
        }
    }

    // ---------------- score = relu(h2 + b1) . w_score, then half-wave reduce ----------------
    float sc[8];
    #pragma unroll
    for (int r = 0; r < 8; ++r) sc[r] = 0.0f;
    #pragma unroll
    for (int nt = 0; nt < 8; ++nt) {
        const int n = nt * 16 + lm;
        const float bb = b1[n];
        const float ww = wsc[n];
        #pragma unroll
        for (int r = 0; r < 8; ++r) {
            float h = fmaxf(acc2[nt][r] + bb, 0.0f);
            sc[r] = fmaf(h, ww, sc[r]);
        }
    }
    #pragma unroll
    for (int r = 0; r < 8; ++r) {
        float t = sc[r];
        t += __shfl_xor(t, 1, 32);
        t += __shfl_xor(t, 2, 32);
        t += __shfl_xor(t, 4, 32);
        t += __shfl_xor(t, 8, 32);     // stays within each 16-lane half
        sc[r] = t;
    }
    if (lm == 0) {                      // lane 0 -> rows m=0..7, lane 16 -> rows m=8..15
        #pragma unroll
        for (int r = 0; r < 8; ++r) {
            const int slot = base + hh * 8 + r;
            if (slot < cnt) scores[list[slot]] = sc[r];
        }
    }
}

// ---------------- deterministic softmax reductions ----------------
__global__ void k_rmax1(const float* __restrict__ s, float* __restrict__ pmax, int M) {
    __shared__ float sm[256];
    int tid = threadIdx.x;
    float m = -3.0e38f;
    for (int i = blockIdx.x * 256 + tid; i < M; i += gridDim.x * 256) m = fmaxf(m, s[i]);
    sm[tid] = m; __syncthreads();
    for (int k = 128; k > 0; k >>= 1) { if (tid < k) sm[tid] = fmaxf(sm[tid], sm[tid + k]); __syncthreads(); }
    if (tid == 0) pmax[blockIdx.x] = sm[0];
}
__global__ void k_rmax2(const float* __restrict__ pmax, float* __restrict__ gmax, int B) {
    __shared__ float sm[256];
    int tid = threadIdx.x;
    sm[tid] = (tid < B) ? pmax[tid] : -3.0e38f; __syncthreads();
    for (int k = 128; k > 0; k >>= 1) { if (tid < k) sm[tid] = fmaxf(sm[tid], sm[tid + k]); __syncthreads(); }
    if (tid == 0) *gmax = sm[0];
}
__global__ void k_rsum1(const float* __restrict__ s, const float* __restrict__ gmax,
                        float* __restrict__ psum, int M) {
    __shared__ float sm[256];
    const float g = *gmax;
    int tid = threadIdx.x;
    float a = 0.0f;
    for (int i = blockIdx.x * 256 + tid; i < M; i += gridDim.x * 256) a += expf(s[i] - g);
    sm[tid] = a; __syncthreads();
    for (int k = 128; k > 0; k >>= 1) { if (tid < k) sm[tid] += sm[tid + k]; __syncthreads(); }
    if (tid == 0) psum[blockIdx.x] = sm[0];
}
__global__ void k_rsum2(const float* __restrict__ psum, float* __restrict__ gsum, int B) {
    __shared__ float sm[256];
    int tid = threadIdx.x;
    sm[tid] = (tid < B) ? psum[tid] : 0.0f; __syncthreads();
    for (int k = 128; k > 0; k >>= 1) { if (tid < k) sm[tid] += sm[tid + k]; __syncthreads(); }
    if (tid == 0) *gsum = sm[0];
}
__global__ void k_out(const float* __restrict__ s, const float* __restrict__ gmax,
                      const float* __restrict__ gsum, float* __restrict__ out, int M) {
    int i = blockIdx.x * blockDim.x + threadIdx.x;
    if (i < M) out[i] = expf(s[i] - *gmax) / *gsum;   // masked rows: exp(-1e30-max)=0
}

extern "C" void kernel_launch(void* const* d_in, const int* in_sizes, int n_in,
                              void* d_out, int out_size, void* d_ws, size_t ws_size,
                              hipStream_t stream) {
    const int*   adj  = (const int*)  d_in[0];
    const float* feat = (const float*)d_in[1];
    const float* W0   = (const float*)d_in[2];
    const float* b0   = (const float*)d_in[3];
    const float* W1   = (const float*)d_in[4];
    const float* b1   = (const float*)d_in[5];
    const float* wsc  = (const float*)d_in[6];
    const int N = in_sizes[0];     // 200000
    const int M = N + 1;

    char* ws = (char*)d_ws;
    size_t off = 0;
    auto alloc = [&](size_t bytes) -> char* {
        char* p = ws + off;
        off = (off + bytes + 63) & ~(size_t)63;
        return p;
    };
    float*          scores  = (float*)alloc((size_t)M * 4);
    int*            list    = (int*)  alloc((size_t)M * 4);
    float*          c0      = (float*)alloc(L0 * 4);
    unsigned short* P0      = (unsigned short*)alloc((size_t)NF * L0 * 2);
    unsigned short* P1      = (unsigned short*)alloc((size_t)L0 * L1 * 2);
    float*          pmax    = (float*)alloc(256 * 4);
    float*          psum    = (float*)alloc(256 * 4);
    float*          gmax    = (float*)alloc(4);
    float*          gsum    = (float*)alloc(4);
    int*            counter = (int*)  alloc(4);

    k_init   <<<(M + 255) / 256, 256, 0, stream>>>(scores, counter, M);
    k_compact<<<(M + 255) / 256, 256, 0, stream>>>(adj, list, counter, N);
    k_c0     <<<1, 256, 0, stream>>>(feat, W0, b0, c0);
    k_pack0  <<<(NF * L0) / 256, 256, 0, stream>>>(W0, P0);
    k_pack1  <<<(L0 * L1) / 256, 256, 0, stream>>>(W1, P1);

    const int tiles  = (M + 15) / 16;        // worst case: all rows survive
    const int blocks = (tiles + 7) / 8;      // 8 waves (tiles) per 256-thread block
    k_mlp<<<blocks, 256, 0, stream>>>(feat, c0, b1, wsc, P0, P1, list, counter, scores, N);

    k_rmax1<<<256, 256, 0, stream>>>(scores, pmax, M);
    k_rmax2<<<1,   256, 0, stream>>>(pmax, gmax, 256);
    k_rsum1<<<256, 256, 0, stream>>>(scores, gmax, psum, M);
    k_rsum2<<<1,   256, 0, stream>>>(psum, gsum, 256);
    k_out  <<<(M + 255) / 256, 256, 0, stream>>>(scores, gmax, gsum, (float*)d_out, M);
}
